// Linear8bit_91001767068465
// MI455X (gfx1250) — 8-bit matrix operand order repaired and checked by reading, compiled, not run on hardware
//
#include <hip/hip_runtime.h>
#include <cstdint>

typedef int v8i __attribute__((ext_vector_type(8)));
typedef int v4i __attribute__((ext_vector_type(4)));
typedef unsigned long long u64;

#define GLOBAL __attribute__((address_space(1)))
#define LOCAL  __attribute__((address_space(3)))

#define TILE_K   64
#define WAVE_M   64
#define WAVE_N   64
#define BLOCK_M  128
#define BLOCK_N  256
#define LDS_STRIDE 80
#define A_BUF_BYTES (BLOCK_M * LDS_STRIDE)
#define B_BUF_BYTES (BLOCK_N * LDS_STRIDE)

__device__ __forceinline__ void async_copy_b128(const GLOBAL int8_t* g, LOCAL int8_t* l) {
    __builtin_amdgcn_global_load_async_to_lds_b128((GLOBAL v4i*)g, (LOCAL v4i*)l, 0, 0);
}

__device__ __forceinline__ v8i lds_a_frag(const LOCAL int8_t* p) {
    union { v8i v; u64 q[4]; } ua;
    ua.q[0] = *(const LOCAL u64*)(p +  0);
    ua.q[1] = *(const LOCAL u64*)(p +  8);
    ua.q[2] = *(const LOCAL u64*)(p + 32);
    ua.q[3] = *(const LOCAL u64*)(p + 40);
    return ua.v;
}

__device__ __forceinline__ v8i lds_b_frag(const LOCAL int8_t* p) {
    union { v8i v; v4i h[2]; } ub;
    ub.h[0] = *(const LOCAL v4i*)(p +  0);
    ub.h[1] = *(const LOCAL v4i*)(p + 32);
    return ub.v;
}

__global__ __launch_bounds__(256)
void int8_gemm_dequant_kernel(const int8_t* __restrict__ xq_,
                              const float*  __restrict__ sx_,
                              const int8_t* __restrict__ wt_,
                              const float*  __restrict__ wsc_,
                              const float*  __restrict__ bias_,
                              _Float16*     __restrict__ out_,
                              int K, int N)
{
    __shared__ int8_t smem[2 * (A_BUF_BYTES + B_BUF_BYTES)];

    const GLOBAL int8_t* xq   = (const GLOBAL int8_t*)xq_;
    const GLOBAL float*  sx   = (const GLOBAL float*)sx_;
    const GLOBAL int8_t* wt   = (const GLOBAL int8_t*)wt_;
    const GLOBAL float*  wsc  = (const GLOBAL float*)wsc_;
    const GLOBAL float*  bias = (const GLOBAL float*)bias_;
    GLOBAL _Float16*     out  = (GLOBAL _Float16*)out_;

    const int tid  = threadIdx.x;
    const int lane = tid & 31;
    const int wave = tid >> 5;
    const int half = lane >> 4;
    const int l16  = lane & 15;
    const int wave_m = wave >> 2;
    const int wave_n = wave & 3;

    const int rowBlk  = blockIdx.y * BLOCK_M;
    const int colBlk  = blockIdx.x * BLOCK_N;
    const int rowBase = rowBlk + wave_m * WAVE_M;
    const int colBase = colBlk + wave_n * WAVE_N;

    const int crow = tid >> 2;
    const int ccol = (tid & 3) * 16;

    const GLOBAL int8_t* gA = xq + (size_t)(rowBlk + crow) * K + ccol;
    const GLOBAL int8_t* gB = wt + (size_t)(colBlk + crow) * K + ccol;

    LOCAL int8_t* ldsBase = (LOCAL int8_t*)smem;
    const int copyOff = crow * LDS_STRIDE + ccol;
    LOCAL int8_t* lA[2] = { ldsBase + copyOff,
                            ldsBase + A_BUF_BYTES + copyOff };
    LOCAL int8_t* lB[2] = { ldsBase + 2 * A_BUF_BYTES + copyOff,
                            ldsBase + 2 * A_BUF_BYTES + B_BUF_BYTES + copyOff };

    const int fragAOff = (wave_m * WAVE_M + l16) * LDS_STRIDE + half * 16;
    const int fragBOff = (wave_n * WAVE_N + l16) * LDS_STRIDE + half * 16;
    const LOCAL int8_t* pA[2] = { ldsBase + fragAOff,
                                  ldsBase + A_BUF_BYTES + fragAOff };
    const LOCAL int8_t* pB[2] = { ldsBase + 2 * A_BUF_BYTES + fragBOff,
                                  ldsBase + 2 * A_BUF_BYTES + B_BUF_BYTES + fragBOff };

    const int T = K / TILE_K;

    auto issue = [&](int buf, int kofs) {
        const GLOBAL int8_t* ga = gA + kofs;
        const GLOBAL int8_t* gb = gB + kofs;
        async_copy_b128(ga,                  lA[buf]);
        async_copy_b128(ga + (size_t)64 * K, lA[buf] + 64 * LDS_STRIDE);
#pragma unroll
        for (int it = 0; it < 4; ++it)
            async_copy_b128(gb + (size_t)(it * 64) * K, lB[buf] + it * 64 * LDS_STRIDE);
    };

    v8i acc[4][4] = {};

    auto kstep = [&](int buf, int t) {
        __builtin_amdgcn_s_wait_asynccnt(0);
        __syncthreads();
        if (t + 1 < T) issue(buf ^ 1, (t + 1) * TILE_K);

        v8i a[4], b[4];
#pragma unroll
        for (int i = 0; i < 4; ++i) a[i] = lds_a_frag(pA[buf] + i * 16 * LDS_STRIDE);
#pragma unroll
        for (int j = 0; j < 4; ++j) b[j] = lds_b_frag(pB[buf] + j * 16 * LDS_STRIDE);

#pragma unroll
        for (int i = 0; i < 4; ++i) {
#pragma unroll
            for (int j = 0; j < 4; ++j) {
                acc[i][j] = __builtin_amdgcn_wmma_i32_16x16x64_iu8(
                    true, a[i], true, b[j], acc[i][j], false, false);
            }
        }
    };

    issue(0, 0);
    for (int t = 0; t < T; t += 2) {
        kstep(0, t);
        kstep(1, t + 1);
    }

#pragma unroll
    for (int i = 0; i < 4; ++i) {
        const int rbase = rowBase + i * 16 + half * 8;
        float s[8];
#pragma unroll
        for (int r = 0; r < 8; ++r) s[r] = sx[rbase + r];
#pragma unroll
        for (int j = 0; j < 4; ++j) {
            const int n  = colBase + j * 16 + l16;
            const float ws = wsc[n];
            const float bb = bias[n];
#pragma unroll
            for (int r = 0; r < 8; ++r) {
                const float v = (float)acc[i][j][r] * s[r] * ws + bb;
                out[(size_t)(rbase + r) * N + n] = (_Float16)v;
            }
        }
    }
}

extern "C" void kernel_launch(void* const* d_in, const int* in_sizes, int n_in,
                              void* d_out, int out_size, void* d_ws, size_t ws_size,
                              hipStream_t stream) {
    const int8_t* xq   = (const int8_t*)d_in[0];
    const float*  sx   = (const float*) d_in[1];
    const int8_t* wt   = (const int8_t*)d_in[2];
    const float*  wsc  = (const float*) d_in[3];
    const float*  bias = (const float*) d_in[4];
    _Float16* out = (_Float16*)d_out;

    const int Mrows = in_sizes[1];
    const int N     = in_sizes[4];
    const int K     = in_sizes[0] / Mrows;

    dim3 grid(N / BLOCK_N, Mrows / BLOCK_M);
    dim3 block(256);
    hipLaunchKernelGGL(int8_gemm_dequant_kernel, grid, block, 0, stream,
                       xq, sx, wt, wsc, bias, out, K, N);
}
